// Elmo_64939905515802
// MI455X (gfx1250) — matrix operand order repaired and checked by reading, not compiled, not run on hardware
//
#include <hip/hip_runtime.h>
#include <hip/hip_bf16.h>
#include <math.h>

// ---------------- model dims (must match reference) ----------------
#define B_     32
#define T_     256
#define NTOK   (B_ * T_)        // 8192
#define MAXC   50
#define CEMB   16
#define NFILT  2048
#define OUTD   512
#define HID    512
#define CELLSZ 4096
#define G4     (4 * CELLSZ)     // 16384
#define CLIPV  3.0f

// ---------------- WMMA types ----------------
typedef __attribute__((ext_vector_type(16))) __bf16 bf16x16;
typedef __attribute__((ext_vector_type(8)))  __bf16 bf16x8;
typedef __attribute__((ext_vector_type(8)))  float  f32x8;

__device__ inline f32x8 zero8() {
    f32x8 z = {0.f, 0.f, 0.f, 0.f, 0.f, 0.f, 0.f, 0.f};
    return z;
}

__device__ inline f32x8 wmma_bf16(bf16x16 a, bf16x16 b, f32x8 c) {
    // v_wmma_f32_16x16x32_bf16 : D = A(16x32) * B(32x16) + C(16x16 f32)
    return __builtin_amdgcn_wmma_f32_16x16x32_bf16(
        false, a, false, b, (short)0, c, false, false);
}

// A fragment: rows m0..m0+15 (lane%16), K k0..k0+31.
// Layout: lanes 0-15 hold K {k0..k0+15},
//         lanes 16-31 hold K {k0+16..k0+31} (matches loadB).
__device__ inline bf16x16 loadA(const __bf16* base, long rowStride, int lane, int k0) {
    int r  = lane & 15;
    int kb = k0 + ((lane >> 4) << 4);
    const __bf16* p = base + (long)r * rowStride + kb;
    return *(const bf16x16*)p;
}

// B fragment from W[N,K] row-major (computing A @ W^T):
// lane n = n0 + lane%16, K contiguous 16 starting at k0 + 16*(lane>=16).
__device__ inline bf16x16 loadB(const __bf16* W, long ldw, int n0, int k0, int lane) {
    const __bf16* p = W + (long)(n0 + (lane & 15)) * ldw + k0 + ((lane >> 4) << 4);
    return *(const bf16x16*)p;
}

__device__ inline float sigmf(float x) { return 1.f / (1.f + __expf(-x)); }
__device__ inline float clip3(float x) { return fminf(fmaxf(x, -CLIPV), CLIPV); }

// ---------------- weight conversion ----------------
__global__ void k_f32_to_bf16(const float* __restrict__ src, __bf16* __restrict__ dst, long n) {
    long i = (long)blockIdx.x * 256 + threadIdx.x;
    if (i < n) dst[i] = (__bf16)src[i];
}

// ---------------- char-CNN encoder (one workgroup per token) ----------------
struct ConvPtrs { const float* W[7]; const float* b[7]; };

__global__ void k_char_encoder(const int* __restrict__ char_ids,
                               const float* __restrict__ char_emb,
                               ConvPtrs cp, __bf16* __restrict__ tok) {
    __shared__ float x[MAXC * CEMB];           // [pos][emb], 3.2 KB
    int tokId = blockIdx.x;
    int tid   = threadIdx.x;
    const int* ids = char_ids + (long)tokId * MAXC;
    for (int i = tid; i < MAXC * CEMB; i += 256) {
        int pos = i / CEMB, e = i % CEMB;
        x[i] = char_emb[(long)ids[pos] * CEMB + e];
    }
    __syncthreads();

    const int cw[7] = {1, 2, 3, 4, 5, 6, 7};
    const int cs[8] = {0, 32, 64, 128, 256, 512, 1024, 2048};
    for (int ch = tid; ch < NFILT; ch += 256) {
        int f = 0;
#pragma unroll
        for (int q = 1; q < 7; ++q) if (ch >= cs[q]) f = q;
        int w = cw[f];
        int o = ch - cs[f];
        const float* Wf = cp.W[f] + (long)o * CEMB * w;   // W[o,i,k] OIH
        int P = MAXC - w + 1;
        float m = -1e30f;
        for (int t = 0; t < P; ++t) {
            float s = 0.f;
            for (int i = 0; i < CEMB; ++i) {
                const float* wr = Wf + i * w;
                for (int k = 0; k < w; ++k)
                    s += x[(t + k) * CEMB + i] * wr[k];
            }
            m = fmaxf(m, s);
        }
        float v = m + cp.b[f][o];
        tok[(long)tokId * NFILT + ch] = (__bf16)fmaxf(v, 0.f);
    }
}

// ---------------- highway layer, fused gate epilogue ----------------
// proj = tok @ W^T + b ; nonlin = proj[:, :2048], gate = sigmoid(proj[:, 2048:])
// tokOut = gate*tok + (1-gate)*relu(nonlin)
__global__ void k_highway(const __bf16* __restrict__ tokIn,
                          const __bf16* __restrict__ W,   // [4096, 2048] bf16
                          const float*  __restrict__ bias, // [4096]
                          __bf16* __restrict__ tokOut) {
    int lane = threadIdx.x & 31;
    int wav  = threadIdx.x >> 5;
    int n0   = (blockIdx.x * 8 + wav) * 16;   // feature column tile (0..2047)
    int m0   = blockIdx.y * 32;               // token row tile
    f32x8 aN0 = zero8(), aN1 = zero8(), aG0 = zero8(), aG1 = zero8();
    for (int k0 = 0; k0 < NFILT; k0 += 32) {
        bf16x16 a0 = loadA(tokIn + (long)m0 * NFILT, NFILT, lane, k0);
        bf16x16 a1 = loadA(tokIn + (long)(m0 + 16) * NFILT, NFILT, lane, k0);
        bf16x16 bn = loadB(W, NFILT, n0, k0, lane);
        bf16x16 bg = loadB(W, NFILT, n0 + NFILT, k0, lane);
        aN0 = wmma_bf16(a0, bn, aN0);
        aN1 = wmma_bf16(a1, bn, aN1);
        aG0 = wmma_bf16(a0, bg, aG0);
        aG1 = wmma_bf16(a1, bg, aG1);
    }
    int n = n0 + (lane & 15);
    float bN = bias[n], bG = bias[n + NFILT];
#pragma unroll
    for (int r = 0; r < 8; ++r) {
        int mA = m0 + ((lane >> 4) << 3) + r;
        int mB = mA + 16;
        float nl = fmaxf(aN0[r] + bN, 0.f);
        float gt = sigmf(aG0[r] + bG);
        float to = (float)tokIn[(long)mA * NFILT + n];
        tokOut[(long)mA * NFILT + n] = (__bf16)(gt * to + (1.f - gt) * nl);
        nl = fmaxf(aN1[r] + bN, 0.f);
        gt = sigmf(aG1[r] + bG);
        to = (float)tokIn[(long)mB * NFILT + n];
        tokOut[(long)mB * NFILT + n] = (__bf16)(gt * to + (1.f - gt) * nl);
    }
}

// ---------------- projection: te = tok @ projW^T + b ----------------
__global__ void k_proj(const __bf16* __restrict__ A,     // [8192, 2048]
                       const __bf16* __restrict__ W,     // [512, 2048]
                       const float*  __restrict__ bias,  // [512]
                       float* __restrict__ outF, __bf16* __restrict__ outB) {
    int lane = threadIdx.x & 31;
    int wav  = threadIdx.x >> 5;
    int n0   = (blockIdx.x * 8 + wav) * 16;
    int m0   = blockIdx.y * 32;
    f32x8 c0 = zero8(), c1 = zero8();
    for (int k0 = 0; k0 < NFILT; k0 += 32) {
        bf16x16 a0 = loadA(A + (long)m0 * NFILT, NFILT, lane, k0);
        bf16x16 a1 = loadA(A + (long)(m0 + 16) * NFILT, NFILT, lane, k0);
        bf16x16 b  = loadB(W, NFILT, n0, k0, lane);
        c0 = wmma_bf16(a0, b, c0);
        c1 = wmma_bf16(a1, b, c1);
    }
    int n = n0 + (lane & 15);
    float bb = bias[n];
#pragma unroll
    for (int r = 0; r < 8; ++r) {
        int mA = m0 + ((lane >> 4) << 3) + r;
        int mB = mA + 16;
        float v = c0[r] + bb;
        outF[(long)mA * OUTD + n] = v;
        outB[(long)mA * OUTD + n] = (__bf16)v;
        v = c1[r] + bb;
        outF[(long)mB * OUTD + n] = v;
        outB[(long)mB * OUTD + n] = (__bf16)v;
    }
}

// ---------------- LSTM step kernel 1: gates = x_t@Wi^T + h@Ws^T + bs ----------------
// X points at the (t) column slice of a [B, T, 512] bf16 sequence (row stride T*512).
__global__ void k_lstm_gates(const __bf16* __restrict__ X,
                             const __bf16* __restrict__ H,   // [32, 512]
                             const __bf16* __restrict__ Wi,  // [16384, 512]
                             const __bf16* __restrict__ Ws,  // [16384, 512]
                             const float*  __restrict__ bs,  // [16384]
                             float* __restrict__ gates) {    // [32, 16384]
    const long xStride = (long)T_ * HID;
    int lane = threadIdx.x & 31;
    int wav  = threadIdx.x >> 5;
    int n0   = (blockIdx.x * 8 + wav) * 16;    // 0..16383
    f32x8 c0 = zero8(), c1 = zero8();
    for (int k0 = 0; k0 < HID; k0 += 32) {
        bf16x16 a0 = loadA(X, xStride, lane, k0);
        bf16x16 a1 = loadA(X + 16 * xStride, xStride, lane, k0);
        bf16x16 b  = loadB(Wi, HID, n0, k0, lane);
        c0 = wmma_bf16(a0, b, c0);
        c1 = wmma_bf16(a1, b, c1);
    }
    for (int k0 = 0; k0 < HID; k0 += 32) {
        bf16x16 a0 = loadA(H, HID, lane, k0);
        bf16x16 a1 = loadA(H + 16 * HID, HID, lane, k0);
        bf16x16 b  = loadB(Ws, HID, n0, k0, lane);
        c0 = wmma_bf16(a0, b, c0);
        c1 = wmma_bf16(a1, b, c1);
    }
    int n = n0 + (lane & 15);
    float bb = bs[n];
#pragma unroll
    for (int r = 0; r < 8; ++r) {
        int m = ((lane >> 4) << 3) + r;
        gates[(long)m * G4 + n]        = c0[r] + bb;
        gates[(long)(m + 16) * G4 + n] = c1[r] + bb;
    }
}

// ---------------- LSTM step kernel 2: cell nonlinearity ----------------
__global__ void k_lstm_cell(const float* __restrict__ gates,
                            float* __restrict__ c, __bf16* __restrict__ hpre) {
    long i = (long)blockIdx.x * 256 + threadIdx.x;     // [0, 32*4096)
    if (i >= (long)B_ * CELLSZ) return;
    int b = (int)(i / CELLSZ), j = (int)(i % CELLSZ);
    const float* g = gates + (long)b * G4;
    float ig = sigmf(g[j]);
    float fg = sigmf(g[CELLSZ + j]);
    float gg = tanhf(g[2 * CELLSZ + j]);
    float og = sigmf(g[3 * CELLSZ + j]);
    float cn = clip3(ig * gg + fg * c[i]);
    c[i] = cn;
    hpre[i] = (__bf16)(og * tanhf(cn));
}

// ---------------- LSTM step kernel 3: h = clip(hpre @ Wp^T) ----------------
__global__ void k_lstm_proj(const __bf16* __restrict__ hpre,  // [32, 4096]
                            const __bf16* __restrict__ Wp,    // [512, 4096]
                            __bf16* __restrict__ h,           // [32, 512] recurrent
                            float* __restrict__ seqF,         // (t) slice of [B,T,512]
                            __bf16* __restrict__ seqB) {      // optional bf16 slice
    const long sStride = (long)T_ * HID;
    int lane = threadIdx.x & 31;
    int wav  = threadIdx.x >> 5;
    int n0   = (blockIdx.x * 8 + wav) * 16;    // 0..511
    f32x8 c0 = zero8(), c1 = zero8();
    for (int k0 = 0; k0 < CELLSZ; k0 += 32) {
        bf16x16 a0 = loadA(hpre, CELLSZ, lane, k0);
        bf16x16 a1 = loadA(hpre + 16 * CELLSZ, CELLSZ, lane, k0);
        bf16x16 b  = loadB(Wp, CELLSZ, n0, k0, lane);
        c0 = wmma_bf16(a0, b, c0);
        c1 = wmma_bf16(a1, b, c1);
    }
    int n = n0 + (lane & 15);
#pragma unroll
    for (int r = 0; r < 8; ++r) {
        int mA = ((lane >> 4) << 3) + r;
        int mB = mA + 16;
        float v = clip3(c0[r]);
        h[(long)mA * HID + n] = (__bf16)v;
        seqF[(long)mA * sStride + n] = v;
        if (seqB) seqB[(long)mA * sStride + n] = (__bf16)v;
        v = clip3(c1[r]);
        h[(long)mB * HID + n] = (__bf16)v;
        seqF[(long)mB * sStride + n] = v;
        if (seqB) seqB[(long)mB * sStride + n] = (__bf16)v;
    }
}

// ---------------- scalar mix ----------------
__global__ void k_mix(const float* __restrict__ te,
                      const float* __restrict__ f0, const float* __restrict__ b0,
                      const float* __restrict__ f1, const float* __restrict__ b1,
                      const float* __restrict__ mixw, const float* __restrict__ gamma,
                      float* __restrict__ out) {
    long i = (long)blockIdx.x * 256 + threadIdx.x;
    if (i >= (long)NTOK * 1024) return;
    float w0 = mixw[0], w1 = mixw[1], w2 = mixw[2];
    float mx = fmaxf(w0, fmaxf(w1, w2));
    float e0 = __expf(w0 - mx), e1 = __expf(w1 - mx), e2 = __expf(w2 - mx);
    float inv = 1.f / (e0 + e1 + e2);
    e0 *= inv; e1 *= inv; e2 *= inv;
    long r = i >> 10;
    int  d = (int)(i & 1023);
    float v;
    if (d < 512) {
        long p = r * 512 + d;
        v = e0 * te[p] + e1 * f0[p] + e2 * (f1[p] + f0[p]);   // layer1 residual
    } else {
        long p = r * 512 + (d - 512);
        v = e0 * te[p] + e1 * b0[p] + e2 * (b1[p] + b0[p]);
    }
    out[i] = gamma[0] * v;
}

// ---------------- host ----------------
extern "C" void kernel_launch(void* const* d_in, const int* in_sizes, int n_in,
                              void* d_out, int out_size, void* d_ws, size_t ws_size,
                              hipStream_t stream) {
    (void)in_sizes; (void)n_in; (void)out_size; (void)ws_size;
    char* ws = (char*)d_ws;
    size_t off = 0;
    auto alloc = [&](size_t bytes) -> char* {
        off = (off + 255) & ~(size_t)255;
        char* p = ws + off;
        off += bytes;
        return p;
    };
    // workspace layout (~366 MB total)
    __bf16* tok0   = (__bf16*)alloc((size_t)NTOK * NFILT * 2);
    __bf16* tok1   = (__bf16*)alloc((size_t)NTOK * NFILT * 2);
    __bf16* hwWb0  = (__bf16*)alloc((size_t)2 * NFILT * NFILT * 2);
    __bf16* hwWb1  = (__bf16*)alloc((size_t)2 * NFILT * NFILT * 2);
    __bf16* projWb = (__bf16*)alloc((size_t)OUTD * NFILT * 2);
    float*  teF    = (float*) alloc((size_t)NTOK * OUTD * 4);
    __bf16* teB    = (__bf16*)alloc((size_t)NTOK * OUTD * 2);
    __bf16* WiB[4]; __bf16* WsB[4]; __bf16* WpB[4];
    for (int dl = 0; dl < 4; ++dl) {
        WiB[dl] = (__bf16*)alloc((size_t)G4 * HID * 2);
        WsB[dl] = (__bf16*)alloc((size_t)G4 * HID * 2);
        WpB[dl] = (__bf16*)alloc((size_t)HID * CELLSZ * 2);
    }
    float* seqF[4];
    for (int dl = 0; dl < 4; ++dl) seqF[dl] = (float*)alloc((size_t)NTOK * HID * 4);
    __bf16* seqBf[2];
    seqBf[0] = (__bf16*)alloc((size_t)NTOK * HID * 2);
    seqBf[1] = (__bf16*)alloc((size_t)NTOK * HID * 2);
    float*  gates = (float*) alloc((size_t)B_ * G4 * 4);
    float*  cbuf  = (float*) alloc((size_t)B_ * CELLSZ * 4);
    __bf16* hpre  = (__bf16*)alloc((size_t)B_ * CELLSZ * 2);
    __bf16* hbuf  = (__bf16*)alloc((size_t)B_ * HID * 2);

    auto cvt = [&](const void* src, __bf16* dst, long n) {
        k_f32_to_bf16<<<dim3((unsigned)((n + 255) / 256)), 256, 0, stream>>>(
            (const float*)src, dst, n);
    };

    // ---- weight conversion to bf16 ----
    cvt(d_in[16], hwWb0, (long)2 * NFILT * NFILT);
    cvt(d_in[17], hwWb1, (long)2 * NFILT * NFILT);
    cvt(d_in[20], projWb, (long)OUTD * NFILT);
    const float* bsF[4];
    for (int dl = 0; dl < 4; ++dl) {
        int base = 22 + dl * 4;                 // Wi, Ws, bs, Wp
        cvt(d_in[base + 0], WiB[dl], (long)G4 * HID);
        cvt(d_in[base + 1], WsB[dl], (long)G4 * HID);
        cvt(d_in[base + 3], WpB[dl], (long)HID * CELLSZ);
        bsF[dl] = (const float*)d_in[base + 2];
    }

    // ---- char encoder ----
    ConvPtrs cp;
    for (int f = 0; f < 7; ++f) {
        cp.W[f] = (const float*)d_in[2 + f];
        cp.b[f] = (const float*)d_in[9 + f];
    }
    k_char_encoder<<<dim3(NTOK), 256, 0, stream>>>(
        (const int*)d_in[0], (const float*)d_in[1], cp, tok0);

    // ---- highway x2 (ping-pong), projection ----
    k_highway<<<dim3(16, 256), 256, 0, stream>>>(tok0, hwWb0, (const float*)d_in[18], tok1);
    k_highway<<<dim3(16, 256), 256, 0, stream>>>(tok1, hwWb1, (const float*)d_in[19], tok0);
    k_proj<<<dim3(4, 256), 256, 0, stream>>>(tok0, projWb, (const float*)d_in[21], teF, teB);

    // ---- biLSTM: dl = {l0 fwd, l0 bwd, l1 fwd, l1 bwd} ----
    const __bf16* lstmIn[4] = { teB, teB, seqBf[0], seqBf[1] };
    __bf16*       outBf[4]  = { seqBf[0], seqBf[1], nullptr, nullptr };
    for (int dl = 0; dl < 4; ++dl) {
        int dir = dl & 1;                       // 0 = fwd, 1 = bwd
        hipMemsetAsync(cbuf, 0, (size_t)B_ * CELLSZ * 4, stream);
        hipMemsetAsync(hbuf, 0, (size_t)B_ * HID * 2, stream);
        for (int s = 0; s < T_; ++s) {
            int t = (dir == 0) ? s : (T_ - 1 - s);
            const __bf16* X = lstmIn[dl] + (size_t)t * HID;
            k_lstm_gates<<<dim3(128), 256, 0, stream>>>(
                X, hbuf, WiB[dl], WsB[dl], bsF[dl], gates);
            k_lstm_cell<<<dim3(512), 256, 0, stream>>>(gates, cbuf, hpre);
            k_lstm_proj<<<dim3(4), 256, 0, stream>>>(
                hpre, WpB[dl], hbuf,
                seqF[dl] + (size_t)t * HID,
                outBf[dl] ? outBf[dl] + (size_t)t * HID : nullptr);
        }
    }

    // ---- scalar mix ----
    long total = (long)NTOK * 1024;
    k_mix<<<dim3((unsigned)((total + 255) / 256)), 256, 0, stream>>>(
        teF, seqF[0], seqF[1], seqF[2], seqF[3],
        (const float*)d_in[38], (const float*)d_in[39], (float*)d_out);
}
